// AttentionLayer_17179869507
// MI455X (gfx1250) — matrix operand order repaired and checked by reading, not compiled, not run on hardware
//
#include <hip/hip_runtime.h>
#include <hip/hip_bf16.h>
#include <math.h>

// Problem constants (match reference)
#define BB 4
#define TQ 128
#define TK 512
#define NN 1024
#define QS 1024
#define VS 1024
#define OS 1024

typedef __attribute__((ext_vector_type(16))) __bf16 v16bf;
typedef __attribute__((ext_vector_type(8)))  float  v8f;

// Hardware V_TANH_F32 (CDNA5 TRANS op) when the builtin exists.
__device__ __forceinline__ float dev_tanh(float x) {
#if __has_builtin(__builtin_amdgcn_tanhf)
    return __builtin_amdgcn_tanhf(x);
#elif __has_builtin(__builtin_amdgcn_tanh_f32)
    return __builtin_amdgcn_tanh_f32(x);
#else
    return tanhf(x);
#endif
}

// ---------- WMMA fragment helpers (bf16 A/B, f32 C) ----------
// A-matrix 16x32 (MxK), bf16, row-major source with leading dim lda (floats).
// lanes 0-15: M=lane, elems 0..15 -> K=kbase+0..15
// lanes16-31: M=lane-16, elems 0..15 -> K=kbase+16..31
__device__ __forceinline__ v16bf frag_a_rowmajor(const float* A, int lda,
                                                 int mbase, int kbase, int lane) {
    int row   = mbase + (lane & 15);
    int khalf = lane >> 4;                // 0 or 1
    const float* p0 = A + (size_t)row * lda + kbase + khalf * 16;
    const float* p1 = p0 + 8;
    v16bf a;
#pragma unroll
    for (int e = 0; e < 8; ++e) a[e]     = (__bf16)p0[e];
#pragma unroll
    for (int e = 0; e < 8; ++e) a[8 + e] = (__bf16)p1[e];
    return a;
}

// B-matrix 32x16 (KxN) from weight W stored [Nout, K] row-major (contiguous K).
// lanes 0-15: N=nbase+lane, K=kbase+0..15 ; lanes 16-31: same N, K=kbase+16..31
__device__ __forceinline__ v16bf frag_b_from_wrow(const float* W, int ldw,
                                                  int nbase, int kbase, int lane) {
    int col  = nbase + (lane & 15);
    int kgrp = lane >> 4;
    const float* p = W + (size_t)col * ldw + kbase + kgrp * 16;
    v16bf b;
#pragma unroll
    for (int e = 0; e < 16; ++e) b[e] = (__bf16)p[e];
    return b;
}

// B-matrix 32x16 (KxN) from V stored [K, Nout] row-major (stride ldv between K).
__device__ __forceinline__ v16bf frag_b_kmajor(const float* V, int ldv,
                                               int nbase, int kbase, int lane) {
    int col  = nbase + (lane & 15);
    int kgrp = lane >> 4;
    const float* p = V + (size_t)(kbase + kgrp * 16) * ldv + col;
    v16bf b;
#pragma unroll
    for (int e = 0; e < 16; ++e) b[e] = (__bf16)p[(size_t)e * ldv];
    return b;
}

template <bool HAS_BIAS, bool DO_TANH>
__device__ __forceinline__ void store_tile(float* C, int ldc, int mbase, int nbase,
                                           int lane, v8f acc, const float* bias) {
    int col   = nbase + (lane & 15);
    int khalf = lane >> 4;
    float bv  = HAS_BIAS ? bias[col] : 0.0f;
#pragma unroll
    for (int v = 0; v < 8; ++v) {
        int m   = mbase + v + 8 * khalf;
        float x = acc[v];
        if (HAS_BIAS) x += bv;          // compile-time guarded: no dead +0.0 adds
        if (DO_TANH)  x = dev_tanh(x);
        C[(size_t)m * ldc + col] = x;
    }
}

// ---------- Generic GEMM: C[M,N] = A[M,K] @ W[N,K]^T (+bias)(+tanh) ----------
// Each wave computes a 16x64 strip: one A fragment feeds 4 WMMAs per k-step.
template <bool HAS_BIAS, bool DO_TANH>
__global__ void gemm_wrow_kernel(const float* __restrict__ A,
                                 const float* __restrict__ W,
                                 const float* __restrict__ bias,
                                 float* __restrict__ C,
                                 int M, int N, int K) {
    int lane = threadIdx.x & 31;
    int wave = threadIdx.x >> 5;
    int task = blockIdx.x * (blockDim.x >> 5) + wave;
    int nGrps = N >> 6;                          // groups of 4 n-tiles
    int total = (M >> 4) * nGrps;
    if (task >= total) return;                   // wave-uniform; EXEC stays full
    int mbase = (task / nGrps) * 16;
    int nbase = (task % nGrps) * 64;

    v8f acc0 = {}, acc1 = {}, acc2 = {}, acc3 = {};
    for (int kb = 0; kb < K; kb += 32) {
        v16bf a  = frag_a_rowmajor(A, K, mbase, kb, lane);
        v16bf b0 = frag_b_from_wrow(W, K, nbase +  0, kb, lane);
        v16bf b1 = frag_b_from_wrow(W, K, nbase + 16, kb, lane);
        v16bf b2 = frag_b_from_wrow(W, K, nbase + 32, kb, lane);
        v16bf b3 = frag_b_from_wrow(W, K, nbase + 48, kb, lane);
        acc0 = __builtin_amdgcn_wmma_f32_16x16x32_bf16(false, a, false, b0, (short)0, acc0, false, false);
        acc1 = __builtin_amdgcn_wmma_f32_16x16x32_bf16(false, a, false, b1, (short)0, acc1, false, false);
        acc2 = __builtin_amdgcn_wmma_f32_16x16x32_bf16(false, a, false, b2, (short)0, acc2, false, false);
        acc3 = __builtin_amdgcn_wmma_f32_16x16x32_bf16(false, a, false, b3, (short)0, acc3, false, false);
    }
    store_tile<HAS_BIAS, DO_TANH>(C, N, mbase, nbase +  0, lane, acc0, bias);
    store_tile<HAS_BIAS, DO_TANH>(C, N, mbase, nbase + 16, lane, acc1, bias);
    store_tile<HAS_BIAS, DO_TANH>(C, N, mbase, nbase + 32, lane, acc2, bias);
    store_tile<HAS_BIAS, DO_TANH>(C, N, mbase, nbase + 48, lane, acc3, bias);
}

// ---------- Bahdanau score + fused softmax ----------
// One block per (b,q); thread t owns key row k=t.
// attq row + w_att are staged into LDS with the CDNA5 async memory->LDS path:
// one GLOBAL_LOAD_ASYNC_TO_LDS_B128 covers both arrays (lanes 0-255 -> s_aq,
// lanes 256-511 -> s_w), drained with s_wait_asynccnt before the barrier.
__global__ void score_softmax_kernel(const float* __restrict__ attq,  // [B*TQ, N]
                                     const float* __restrict__ keys,  // [B, TK, N]
                                     const float* __restrict__ w_att, // [N]
                                     const float* __restrict__ b_att, // [1]
                                     float* __restrict__ probs) {     // [B*TQ, TK]
    __shared__ float s_aq[NN];
    __shared__ float s_w[NN];
    __shared__ float s_red[TK];

    int bq = blockIdx.x;          // b*TQ + q
    int t  = threadIdx.x;         // k index, blockDim.x == TK
    const float* aq = attq + (size_t)bq * NN;

    // Async stage: 512 lanes x 16B = 8KB = s_aq(4KB) + s_w(4KB)
    {
        const float* gsrc = (t < 256) ? (aq + t * 4) : (w_att + (t - 256) * 4);
        float*       ldst = (t < 256) ? (s_aq + t * 4) : (s_w + (t - 256) * 4);
        unsigned laddr = (unsigned)(uintptr_t)ldst;   // LDS byte offset (low 32b)
        asm volatile("global_load_async_to_lds_b128 %0, %1, off"
                     :: "v"(laddr), "v"(gsrc) : "memory");
        asm volatile("s_wait_asynccnt 0x0" ::: "memory");
    }
    __syncthreads();

    int b = bq / TQ;
    const float* krow = keys + ((size_t)b * TK + t) * NN;
    float acc = 0.0f;
#pragma unroll 4
    for (int n = 0; n < NN; n += 4) {
        float4 kv = *(const float4*)(krow + n);
        acc += dev_tanh(s_aq[n + 0] + kv.x) * s_w[n + 0];
        acc += dev_tanh(s_aq[n + 1] + kv.y) * s_w[n + 1];
        acc += dev_tanh(s_aq[n + 2] + kv.z) * s_w[n + 2];
        acc += dev_tanh(s_aq[n + 3] + kv.w) * s_w[n + 3];
    }
    float score = acc + b_att[0];

    // block-wide max
    s_red[t] = score;
    __syncthreads();
    for (int s = TK >> 1; s > 0; s >>= 1) {
        if (t < s) s_red[t] = fmaxf(s_red[t], s_red[t + s]);
        __syncthreads();
    }
    float mx = s_red[0];
    __syncthreads();

    float e = __expf(score - mx);
    s_red[t] = e;
    __syncthreads();
    for (int s = TK >> 1; s > 0; s >>= 1) {
        if (t < s) s_red[t] += s_red[t + s];
        __syncthreads();
    }
    float sum = s_red[0];
    probs[(size_t)bq * TK + t] = e / sum;
}

// ---------- Batched context = probs @ values (16x64 strip per wave) ----------
__global__ void context_gemm_kernel(const float* __restrict__ P,   // [B, TQ, TK]
                                    const float* __restrict__ V,   // [B, TK, VS]
                                    float* __restrict__ Ctx) {     // [B, TQ, VS]
    int lane = threadIdx.x & 31;
    int wave = threadIdx.x >> 5;
    int task = blockIdx.x * (blockDim.x >> 5) + wave;
    const int tilesM = TQ >> 4, nGrps = VS >> 6;
    const int perB = tilesM * nGrps;
    if (task >= BB * perB) return;
    int b   = task / perB;
    int r   = task % perB;
    int mbase = (r / nGrps) * 16;
    int nbase = (r % nGrps) * 64;

    const float* Pb = P + (size_t)b * TQ * TK;
    const float* Vb = V + (size_t)b * TK * VS;

    v8f acc0 = {}, acc1 = {}, acc2 = {}, acc3 = {};
    for (int kb = 0; kb < TK; kb += 32) {
        v16bf a  = frag_a_rowmajor(Pb, TK, mbase, kb, lane);
        v16bf b0 = frag_b_kmajor(Vb, VS, nbase +  0, kb, lane);
        v16bf b1 = frag_b_kmajor(Vb, VS, nbase + 16, kb, lane);
        v16bf b2 = frag_b_kmajor(Vb, VS, nbase + 32, kb, lane);
        v16bf b3 = frag_b_kmajor(Vb, VS, nbase + 48, kb, lane);
        acc0 = __builtin_amdgcn_wmma_f32_16x16x32_bf16(false, a, false, b0, (short)0, acc0, false, false);
        acc1 = __builtin_amdgcn_wmma_f32_16x16x32_bf16(false, a, false, b1, (short)0, acc1, false, false);
        acc2 = __builtin_amdgcn_wmma_f32_16x16x32_bf16(false, a, false, b2, (short)0, acc2, false, false);
        acc3 = __builtin_amdgcn_wmma_f32_16x16x32_bf16(false, a, false, b3, (short)0, acc3, false, false);
    }
    float* Cb = Ctx + (size_t)b * TQ * VS;
    store_tile<false, false>(Cb, VS, mbase, nbase +  0, lane, acc0, nullptr);
    store_tile<false, false>(Cb, VS, mbase, nbase + 16, lane, acc1, nullptr);
    store_tile<false, false>(Cb, VS, mbase, nbase + 32, lane, acc2, nullptr);
    store_tile<false, false>(Cb, VS, mbase, nbase + 48, lane, acc3, nullptr);
}

// ---------- out = tanh(cat(query, ctx) @ Wout^T + bout) ----------
__global__ void gemm_concat_kernel(const float* __restrict__ A0,   // [M, QS]
                                   const float* __restrict__ A1,   // [M, VS]
                                   const float* __restrict__ W,    // [N, QS+VS]
                                   const float* __restrict__ bias,
                                   float* __restrict__ C,
                                   int M, int N) {
    int lane = threadIdx.x & 31;
    int wave = threadIdx.x >> 5;
    int task = blockIdx.x * (blockDim.x >> 5) + wave;
    int nGrps = N >> 6;
    int total = (M >> 4) * nGrps;
    if (task >= total) return;
    int mbase = (task / nGrps) * 16;
    int nbase = (task % nGrps) * 64;
    const int K = QS + VS;

    v8f acc0 = {}, acc1 = {}, acc2 = {}, acc3 = {};
    for (int kb = 0; kb < K; kb += 32) {
        const float* Asrc = (kb < QS) ? A0 : A1;
        int kk = (kb < QS) ? kb : (kb - QS);
        v16bf a  = frag_a_rowmajor(Asrc, QS, mbase, kk, lane);   // QS == VS
        v16bf b0 = frag_b_from_wrow(W, K, nbase +  0, kb, lane);
        v16bf b1 = frag_b_from_wrow(W, K, nbase + 16, kb, lane);
        v16bf b2 = frag_b_from_wrow(W, K, nbase + 32, kb, lane);
        v16bf b3 = frag_b_from_wrow(W, K, nbase + 48, kb, lane);
        acc0 = __builtin_amdgcn_wmma_f32_16x16x32_bf16(false, a, false, b0, (short)0, acc0, false, false);
        acc1 = __builtin_amdgcn_wmma_f32_16x16x32_bf16(false, a, false, b1, (short)0, acc1, false, false);
        acc2 = __builtin_amdgcn_wmma_f32_16x16x32_bf16(false, a, false, b2, (short)0, acc2, false, false);
        acc3 = __builtin_amdgcn_wmma_f32_16x16x32_bf16(false, a, false, b3, (short)0, acc3, false, false);
    }
    store_tile<true, true>(C, N, mbase, nbase +  0, lane, acc0, bias);
    store_tile<true, true>(C, N, mbase, nbase + 16, lane, acc1, bias);
    store_tile<true, true>(C, N, mbase, nbase + 32, lane, acc2, bias);
    store_tile<true, true>(C, N, mbase, nbase + 48, lane, acc3, bias);
}

extern "C" void kernel_launch(void* const* d_in, const int* in_sizes, int n_in,
                              void* d_out, int out_size, void* d_ws, size_t ws_size,
                              hipStream_t stream) {
    const float* query  = (const float*)d_in[0];   // [B, TQ, QS]
    const float* keys   = (const float*)d_in[1];   // [B, TK, N]
    const float* values = (const float*)d_in[2];   // [B, TK, VS]
    const float* Wq     = (const float*)d_in[3];   // [N, QS]
    const float* bq     = (const float*)d_in[4];   // [N]
    const float* w_att  = (const float*)d_in[5];   // [N]
    const float* b_att  = (const float*)d_in[6];   // [1]
    const float* Wout   = (const float*)d_in[7];   // [OS, QS+VS]
    const float* bout   = (const float*)d_in[8];   // [OS]

    float* out   = (float*)d_out;                         // [B, TQ, OS]
    float* probs = out + (size_t)BB * TQ * OS;            // [B, TQ, TK] (2nd output)
    float* attq  = (float*)d_ws;                          // [B*TQ, N]
    float* ctx   = attq + (size_t)BB * TQ * NN;           // [B*TQ, VS]

    // 1) attq = query @ Wq^T + bq
    {
        int M = BB * TQ, N = NN;
        int tasks = (M >> 4) * (N >> 6);                  // 512 wave-tasks
        gemm_wrow_kernel<true, false><<<(tasks + 7) / 8, 256, 0, stream>>>(
            query, Wq, bq, attq, M, N, QS);
    }
    // 2) scores + softmax -> probs (directly in d_out scores region)
    score_softmax_kernel<<<BB * TQ, TK, 0, stream>>>(attq, keys, w_att, b_att, probs);

    // 3) ctx = probs @ values (batched)
    {
        int tasks = BB * (TQ >> 4) * (VS >> 6);           // 512
        context_gemm_kernel<<<(tasks + 7) / 8, 256, 0, stream>>>(probs, values, ctx);
    }
    // 4) out = tanh(cat(query, ctx) @ Wout^T + bout)
    {
        int M = BB * TQ, N = OS;
        int tasks = (M >> 4) * (N >> 6);                  // 512
        gemm_concat_kernel<<<(tasks + 7) / 8, 256, 0, stream>>>(
            query, ctx, Wout, bout, out, M, N);
    }
}
